// MHSA_14328010900301
// MI455X (gfx1250) — matrix operand order repaired and checked by reading, not compiled, not run on hardware
//
#include <hip/hip_runtime.h>
#include <hip/hip_bf16.h>

// ---------------- problem constants ----------------
constexpr int BATCH = 4;
constexpr int CH    = 320;            // C
constexpr int NSP   = 16 * 16 * 16;   // N = 4096 spatial positions
constexpr int NH    = 2;              // E heads
constexpr int DKQ   = 32;             // key/query dim per head (== one WMMA K step)
constexpr int DVV   = 160;            // value dim per head
constexpr int FD    = NH * DVV;       // 320 merged feature dim
constexpr int BE    = BATCH * NH;     // 8 (batch, head) pairs
constexpr int NKC   = 32;             // keys per attention chunk

typedef __attribute__((ext_vector_type(16))) _Float16 v16h;
typedef __attribute__((ext_vector_type(8)))  float    v8f;

union Frag { v16h h; unsigned int u[8]; };

// 16-bit A-matrix (16x32): VGPR j holds packed K pair; lanes 0-15 K in {0..7,16..23},
// lanes 16-31 shifted by +8 (ISA 7.12.2).
__device__ __forceinline__ int kA16(int j, int half) {
  return (j < 4 ? 2 * j : 16 + 2 * (j - 4)) + 8 * half;
}
// 16-bit B-matrix (32x16): same K order as A; lanes 0-15 K in {0..7,16..23}, lanes 16-31 shifted by +8.
__device__ __forceinline__ int kB16(int j, int half) {
  return (j < 4 ? 2 * j : 16 + 2 * (j - 4)) + 8 * half;
}

// ---------------------------------------------------------------------------
// 1) Pointwise projections: Q/K raw f32 (channel-major), V f16 [dv][n].
//    Each block: 8 output rows x 256 spatial positions -> x loads reused x8.
// ---------------------------------------------------------------------------
__global__ __launch_bounds__(256)
void mhsa_proj_kernel(const float* __restrict__ x,
                      const float* __restrict__ Wq, const float* __restrict__ bq,
                      const float* __restrict__ Wk, const float* __restrict__ bk,
                      const float* __restrict__ Wv, const float* __restrict__ bv,
                      float* __restrict__ Qraw, float* __restrict__ Kraw,
                      _Float16* __restrict__ Vh)
{
  const int n  = blockIdx.x * 256 + threadIdx.x;
  const int r0 = blockIdx.y * 8;          // 0..223 (q:0-31, k:32-63, v:64-223)
  const int be = blockIdx.z;
  const int b  = be >> 1, e = be & 1;

  const float* Wrow;
  const float* bias;
  int seg, rr;
  if (r0 < DKQ)        { seg = 0; rr = r0;           Wrow = Wq + (size_t)(e * DKQ + rr) * CH; bias = bq + e * DKQ + rr; }
  else if (r0 < 2*DKQ) { seg = 1; rr = r0 - DKQ;     Wrow = Wk + (size_t)(e * DKQ + rr) * CH; bias = bk + e * DKQ + rr; }
  else                 { seg = 2; rr = r0 - 2 * DKQ; Wrow = Wv + (size_t)(e * DVV + rr) * CH; bias = bv + e * DVV + rr; }

  const float* xcol = x + (size_t)b * CH * NSP + n;
  float acc[8];
  #pragma unroll
  for (int i = 0; i < 8; ++i) acc[i] = 0.f;

  for (int c = 0; c < CH; ++c) {
    const float xv = xcol[(size_t)c * NSP];            // coalesced across lanes
    #pragma unroll
    for (int i = 0; i < 8; ++i)                        // wave-uniform weights -> s_loads
      acc[i] = fmaf(Wrow[(size_t)i * CH + c], xv, acc[i]);
  }

  if (seg == 0) {
    #pragma unroll
    for (int i = 0; i < 8; ++i)
      Qraw[((size_t)be * DKQ + rr + i) * NSP + n] = acc[i] + bias[i];
  } else if (seg == 1) {
    #pragma unroll
    for (int i = 0; i < 8; ++i)
      Kraw[((size_t)be * DKQ + rr + i) * NSP + n] = acc[i] + bias[i];
  } else {
    #pragma unroll
    for (int i = 0; i < 8; ++i)
      Vh[((size_t)be * DVV + rr + i) * NSP + n] = (_Float16)(acc[i] + bias[i]);
  }
}

// ---------------------------------------------------------------------------
// 2) L2-normalize Q,K over dk; transpose to [n][dk] f16 (b128 packed stores).
// ---------------------------------------------------------------------------
__global__ __launch_bounds__(256)
void mhsa_l2norm_kernel(const float* __restrict__ Qraw, const float* __restrict__ Kraw,
                        _Float16* __restrict__ Qn, _Float16* __restrict__ Kn)
{
  const int idx = blockIdx.x * 256 + threadIdx.x;   // BE*NSP threads
  const int be  = idx >> 12;                        // NSP == 4096
  const int n   = idx & (NSP - 1);

  const float* srcs[2] = { Qraw + (size_t)be * DKQ * NSP + n,
                           Kraw + (size_t)be * DKQ * NSP + n };
  _Float16*    dsts[2] = { Qn + ((size_t)be * NSP + n) * DKQ,
                           Kn + ((size_t)be * NSP + n) * DKQ };

  #pragma unroll
  for (int s = 0; s < 2; ++s) {
    const float* src = srcs[s];
    float v[DKQ]; float ss = 0.f;
    #pragma unroll
    for (int j = 0; j < DKQ; ++j) { v[j] = src[(size_t)j * NSP]; ss = fmaf(v[j], v[j], ss); }
    const float sc = 1.f / fmaxf(sqrtf(ss), 1e-6f);
    uint4* d = (uint4*)dsts[s];
    #pragma unroll
    for (int g = 0; g < 4; ++g) {
      union { _Float16 h[8]; uint4 q; } pk;
      #pragma unroll
      for (int jj = 0; jj < 8; ++jj) pk.h[jj] = (_Float16)(v[g * 8 + jj] * sc);
      d[g] = pk.q;
    }
  }
}

// ---------------------------------------------------------------------------
// 3) Convert merge weights to f16 once.
// ---------------------------------------------------------------------------
__global__ __launch_bounds__(256)
void mhsa_cvt_kernel(const float* __restrict__ Wm, _Float16* __restrict__ Wm16)
{
  const int i = blockIdx.x * 256 + threadIdx.x;
  if (i < CH * FD) Wm16[i] = (_Float16)Wm[i];
}

// ---------------------------------------------------------------------------
// 4) Attention. Q,K are L2-normalized -> scores in [-1,1]; softmax is shift-
//    invariant, so use FIXED shift exp(s-1): no running max, no O rescaling,
//    no per-chunk cross-lane reductions. Each wave owns 32 query rows
//    (2 A-tiles) so every K/V B-fragment read from LDS feeds two WMMAs.
//    Key-permutation trick: the two S-tiles use EVEN/ODD keys, so each lane's
//    (p0,p1) pair packs into one dword -> single ds_store_b32 per row, and
//    Pst lands in natural key order (A-frag reads & V fragments unchanged).
// ---------------------------------------------------------------------------
__global__ __launch_bounds__(256)
void mhsa_attn_kernel(const _Float16* __restrict__ Qn, const _Float16* __restrict__ Kn,
                      const _Float16* __restrict__ Vh, _Float16* __restrict__ Rh)
{
  __shared__ _Float16 Kc[NKC][DKQ];      // 2 KB  [key][dk]
  __shared__ _Float16 Vc[DVV][NKC];      // 10 KB [dv][key]
  __shared__ _Float16 Pst[8][32][NKC];   // 16 KB per-wave P staging [32 q][32 key]

  const int tid  = threadIdx.x;
  const int wv   = tid >> 5;
  const int lane = tid & 31;
  const int half = lane >> 4;
  const int lr   = lane & 15;

  const int be   = blockIdx.x >> 4;      // 16 query blocks of 256 per (b,e)
  const int qblk = blockIdx.x & 15;
  const int q0   = qblk * 256 + wv * 32; // wave owns 32 query rows

  const unsigned int* Qu  = (const unsigned int*)(Qn + ((size_t)be * NSP + q0) * DKQ);
  const unsigned int* Knu = (const unsigned int*)(Kn + (size_t)be * NSP * DKQ);
  const unsigned int* Vu  = (const unsigned int*)(Vh + (size_t)be * DVV * NSP);

  // Two A fragments of Qn (held for the whole key loop)
  Frag aq0, aq1;
  #pragma unroll
  for (int j = 0; j < 8; ++j) {
    const int ki = kA16(j, half) >> 1;
    aq0.u[j] = Qu[lr * (DKQ / 2) + ki];
    aq1.u[j] = Qu[(16 + lr) * (DKQ / 2) + ki];
  }

  v8f zero;
  #pragma unroll
  for (int r = 0; r < 8; ++r) zero[r] = 0.f;

  v8f O0[10], O1[10];
  #pragma unroll
  for (int t = 0; t < 10; ++t) { O0[t] = zero; O1[t] = zero; }

  float ls0[8], ls1[8];                  // lane-private partial softmax sums
  #pragma unroll
  for (int r = 0; r < 8; ++r) { ls0[r] = 0.f; ls1[r] = 0.f; }

  for (int m0 = 0; m0 < NSP; m0 += NKC) {
    // ---- cooperative chunk staging (coalesced) ----
    unsigned int* kd = (unsigned int*)&Kc[0][0];             // 512 dwords
    kd[tid]       = Knu[m0 * (DKQ / 2) + tid];
    kd[tid + 256] = Knu[m0 * (DKQ / 2) + tid + 256];
    unsigned int* vd = (unsigned int*)&Vc[0][0];             // 2560 dwords
    #pragma unroll
    for (int k = 0; k < 10; ++k) {
      const int t  = tid + k * 256;
      const int dv = t >> 4, w2 = t & 15;
      vd[t] = Vu[(size_t)dv * (NSP / 2) + (m0 >> 1) + w2];
    }
    if (m0 + NKC < NSP)
      __builtin_prefetch(Knu + (m0 + NKC) * (DKQ / 2) + tid, 0, 0);   // global_prefetch
    __syncthreads();

    // ---- S tiles: bk0 = EVEN keys, bk1 = ODD keys (B shared by both A tiles)
    Frag bk0, bk1;
    #pragma unroll
    for (int j = 0; j < 8; ++j) {
      const int kb = kB16(j, half);
      bk0.u[j] = *(const unsigned int*)&Kc[2 * lr][kb];
      bk1.u[j] = *(const unsigned int*)&Kc[2 * lr + 1][kb];
    }

    // A-tile 0 (query rows q0..q0+15)
    {
      v8f S0 = __builtin_amdgcn_wmma_f32_16x16x32_f16(false, aq0.h, false, bk0.h, (short)0, zero, false, false);
      v8f S1 = __builtin_amdgcn_wmma_f32_16x16x32_f16(false, aq0.h, false, bk1.h, (short)0, zero, false, false);
      #pragma unroll
      for (int r = 0; r < 8; ++r) {
        const float p0 = __expf(S0[r] - 1.f);   // fixed shift: |q.k| <= 1
        const float p1 = __expf(S1[r] - 1.f);
        ls0[r] += p0 + p1;
        union { _Float16 h[2]; unsigned int u; } pk;
        pk.h[0] = (_Float16)p0;                 // key 2*lr
        pk.h[1] = (_Float16)p1;                 // key 2*lr+1
        ((unsigned int*)&Pst[wv][r + 8 * half][0])[lr] = pk.u;
      }
    }
    // A-tile 1 (query rows q0+16..q0+31)
    {
      v8f S0 = __builtin_amdgcn_wmma_f32_16x16x32_f16(false, aq1.h, false, bk0.h, (short)0, zero, false, false);
      v8f S1 = __builtin_amdgcn_wmma_f32_16x16x32_f16(false, aq1.h, false, bk1.h, (short)0, zero, false, false);
      #pragma unroll
      for (int r = 0; r < 8; ++r) {
        const float p0 = __expf(S0[r] - 1.f);
        const float p1 = __expf(S1[r] - 1.f);
        ls1[r] += p0 + p1;
        union { _Float16 h[2]; unsigned int u; } pk;
        pk.h[0] = (_Float16)p0;
        pk.h[1] = (_Float16)p1;
        ((unsigned int*)&Pst[wv][16 + r + 8 * half][0])[lr] = pk.u;
      }
    }
    __builtin_amdgcn_wave_barrier();
    asm volatile("s_wait_dscnt 0" ::: "memory");   // wave-local LDS RAW

    Frag ap0, ap1;
    #pragma unroll
    for (int j = 0; j < 8; ++j) {
      const int k0 = kA16(j, half);
      ap0.u[j] = *(const unsigned int*)&Pst[wv][lr][k0];
      ap1.u[j] = *(const unsigned int*)&Pst[wv][16 + lr][k0];
    }

    // ---- O += P x V^T : each V B-fragment feeds two WMMAs ----
    #pragma unroll
    for (int t = 0; t < 10; ++t) {
      Frag bvf;
      #pragma unroll
      for (int j = 0; j < 8; ++j)
        bvf.u[j] = *(const unsigned int*)&Vc[t * 16 + lr][kB16(j, half)];
      O0[t] = __builtin_amdgcn_wmma_f32_16x16x32_f16(false, ap0.h, false, bvf.h, (short)0, O0[t], false, false);
      O1[t] = __builtin_amdgcn_wmma_f32_16x16x32_f16(false, ap1.h, false, bvf.h, (short)0, O1[t], false, false);
    }
    __syncthreads();
  }

  // single cross-lane reduction of the softmax denominators (per 16-lane half)
  float inv0[8], inv1[8];
  #pragma unroll
  for (int r = 0; r < 8; ++r) {
    float a = ls0[r], b2 = ls1[r];
    a  += __shfl_xor(a, 1, 32);  b2 += __shfl_xor(b2, 1, 32);
    a  += __shfl_xor(a, 2, 32);  b2 += __shfl_xor(b2, 2, 32);
    a  += __shfl_xor(a, 4, 32);  b2 += __shfl_xor(b2, 4, 32);
    a  += __shfl_xor(a, 8, 32);  b2 += __shfl_xor(b2, 8, 32);
    inv0[r] = 1.f / fmaxf(a, 1e-20f);
    inv1[r] = 1.f / fmaxf(b2, 1e-20f);
  }

  const int b = be >> 1, e = be & 1;
  #pragma unroll
  for (int t = 0; t < 10; ++t) {
    #pragma unroll
    for (int r = 0; r < 8; ++r) {
      const int f   = e * DVV + t * 16 + lr;
      const int nq0 = q0 + r + 8 * half;
      const int nq1 = nq0 + 16;
      Rh[((size_t)b * NSP + nq0) * FD + f] = (_Float16)(O0[t][r] * inv0[r]);
      Rh[((size_t)b * NSP + nq1) * FD + f] = (_Float16)(O1[t][r] * inv1[r]);
    }
  }
}

// ---------------------------------------------------------------------------
// 5) Merge: out = Wm x Rcat + bm + x. Each wave computes two c-tiles (ct and
//    ct+10) sharing every B fragment of R -> half the B traffic per WMMA.
// ---------------------------------------------------------------------------
__global__ __launch_bounds__(256)
void mhsa_merge_kernel(const _Float16* __restrict__ Wm16, const _Float16* __restrict__ Rh,
                       const float* __restrict__ bm, const float* __restrict__ x,
                       float* __restrict__ out)
{
  const int tid  = threadIdx.x;
  const int wv   = tid >> 5;
  const int lane = tid & 31;
  const int half = lane >> 4;
  const int lr   = lane & 15;

  const int w   = blockIdx.x * 8 + wv;       // (b, ct-pair, nt)
  const int b   = w / (10 * 256);            // 10 c-tile pairs x 256 n-tiles
  const int rem = w % (10 * 256);
  const int ct  = rem >> 8;                  // c-tiles ct and ct+10
  const int nt  = rem & 255;

  const unsigned int* Au = (const unsigned int*)Wm16;
  const unsigned int* Bu = (const unsigned int*)Rh;

  v8f acc0, acc1;
  #pragma unroll
  for (int r = 0; r < 8; ++r) { acc0[r] = 0.f; acc1[r] = 0.f; }

  const int    arow0 = ct * 16 + lr;
  const int    arow1 = (ct + 10) * 16 + lr;
  const size_t brow  = ((size_t)b * NSP + nt * 16 + lr) * FD;

  #pragma unroll
  for (int kk = 0; kk < FD; kk += 32) {
    Frag a0, a1, bb;
    #pragma unroll
    for (int j = 0; j < 8; ++j) {
      const int ka = kA16(j, half);
      a0.u[j] = Au[((size_t)arow0 * FD + kk + ka) >> 1];
      a1.u[j] = Au[((size_t)arow1 * FD + kk + ka) >> 1];
      bb.u[j] = Bu[(brow + kk + kB16(j, half)) >> 1];
    }
    acc0 = __builtin_amdgcn_wmma_f32_16x16x32_f16(false, a0.h, false, bb.h, (short)0, acc0, false, false);
    acc1 = __builtin_amdgcn_wmma_f32_16x16x32_f16(false, a1.h, false, bb.h, (short)0, acc1, false, false);
  }

  #pragma unroll
  for (int r = 0; r < 8; ++r) {
    const int    c0  = ct * 16 + r + 8 * half;
    const int    c1  = c0 + 160;
    const size_t oi0 = ((size_t)b * CH + c0) * NSP + nt * 16 + lr;
    const size_t oi1 = ((size_t)b * CH + c1) * NSP + nt * 16 + lr;
    out[oi0] = acc0[r] + bm[c0] + x[oi0];
    out[oi1] = acc1[r] + bm[c1] + x[oi1];
  }
}

// ---------------------------------------------------------------------------
extern "C" void kernel_launch(void* const* d_in, const int* in_sizes, int n_in,
                              void* d_out, int out_size, void* d_ws, size_t ws_size,
                              hipStream_t stream) {
  const float* x  = (const float*)d_in[0];
  const float* Wq = (const float*)d_in[1];
  const float* bq = (const float*)d_in[2];
  const float* Wk = (const float*)d_in[3];
  const float* bk = (const float*)d_in[4];
  const float* Wv = (const float*)d_in[5];
  const float* bv = (const float*)d_in[6];
  const float* Wm = (const float*)d_in[7];
  const float* bm = (const float*)d_in[8];

  char* ws = (char*)d_ws;
  auto alloc = [&](size_t bytes) -> void* {
    void* p = ws;
    ws += (bytes + 255) & ~(size_t)255;
    return p;
  };
  float*    Qraw = (float*)   alloc((size_t)BE * DKQ * NSP * sizeof(float));     // 4 MB
  float*    Kraw = (float*)   alloc((size_t)BE * DKQ * NSP * sizeof(float));     // 4 MB
  _Float16* Qn   = (_Float16*)alloc((size_t)BE * NSP * DKQ * 2);                 // 2 MB
  _Float16* Kn   = (_Float16*)alloc((size_t)BE * NSP * DKQ * 2);                 // 2 MB
  _Float16* Vh   = (_Float16*)alloc((size_t)BE * DVV * NSP * 2);                 // 10 MB
  _Float16* Rh   = (_Float16*)alloc((size_t)BATCH * NSP * FD * 2);               // 10 MB
  _Float16* Wm16 = (_Float16*)alloc((size_t)CH * FD * 2);                        // 200 KB

  mhsa_proj_kernel<<<dim3(NSP / 256, (2 * DKQ + DVV) / 8, BE), 256, 0, stream>>>(
      x, Wq, bq, Wk, bk, Wv, bv, Qraw, Kraw, Vh);
  mhsa_l2norm_kernel<<<dim3(BE * NSP / 256), 256, 0, stream>>>(Qraw, Kraw, Qn, Kn);
  mhsa_cvt_kernel<<<dim3((CH * FD + 255) / 256), 256, 0, stream>>>(Wm, Wm16);
  mhsa_attn_kernel<<<dim3(BE * (NSP / 256)), 256, 0, stream>>>(Qn, Kn, Vh, Rh);
  mhsa_merge_kernel<<<dim3(BATCH * 10 * 256 / 8), 256, 0, stream>>>(
      Wm16, Rh, bm, x, (float*)d_out);

  (void)in_sizes; (void)n_in; (void)out_size; (void)ws_size;
}
